// GNNLongitudinal_69913477644644
// MI455X (gfx1250) — matrix operand order repaired and checked by reading, not compiled, not run on hardware
//
#include <hip/hip_runtime.h>

#define NN 102400
#define BB 4096
#define EE 1000000
#define TT 10
#define FF 32
#define HH 64
#define DD 128

typedef __bf16 bf16;
typedef __attribute__((ext_vector_type(16))) __bf16 v16bf;
typedef __attribute__((ext_vector_type(8)))  __bf16 v8bf;
typedef __attribute__((ext_vector_type(8)))  float   v8f;
typedef __attribute__((ext_vector_type(4)))  float   v4f;

// ---- fast transcendentals: use CDNA5 V_TANH_F32 / V_RCP_F32 when available ----
__device__ __forceinline__ float fast_tanh(float x) {
#if __has_builtin(__builtin_amdgcn_tanhf)
  return __builtin_amdgcn_tanhf(x);
#elif __has_builtin(__builtin_amdgcn_tanh_f32)
  return __builtin_amdgcn_tanh_f32(x);
#else
  float e = __expf(2.0f * x);
  return (e - 1.0f) / (e + 1.0f);
#endif
}

__device__ __forceinline__ float sigm(float x) {
#if __has_builtin(__builtin_amdgcn_rcpf)
  return __builtin_amdgcn_rcpf(1.0f + __expf(-x));
#else
  return 1.0f / (1.0f + __expf(-x));
#endif
}

__device__ __forceinline__ v8f vzero8() {
  v8f z;
#pragma unroll
  for (int i = 0; i < 8; ++i) z[i] = 0.0f;
  return z;
}

__device__ __forceinline__ v8f wmma_bf16(v16bf a, v16bf b, v8f c) {
  // (neg_a, A, neg_b, B, c_mod, C, reuse_a, reuse_b)
  return __builtin_amdgcn_wmma_f32_16x16x32_bf16(false, a, false, b, (short)0, c, false, false);
}

// A fragment (16x32 bf16): lane = half*16 + M; elem i -> K = half*16 + i
// (16 contiguous bf16).  p points at K=0 of this lane's row.
__device__ __forceinline__ v16bf afrag(const bf16* p, int half) {
  union { v16bf v; v8bf h[2]; } u;
  u.h[0] = *(const v8bf*)(p + half * 16);
  u.h[1] = *(const v8bf*)(p + half * 16 + 8);
  return u.v;
}

// B fragment (32x16 bf16) from column-major (transposed) LDS weight image:
// lane = half*16 + N; elem i -> K = half*16 + i  (16 contiguous bf16).
__device__ __forceinline__ v16bf bfragW(const bf16* sWT, int j, int kb, int half, int kstride) {
  return *(const v16bf*)(sWT + (size_t)j * kstride + kb + half * 16);
}

// ---------------------------------------------------------------------------
// Bidirectional LSTM: x [rows, T=10, F=32] -> out [rows, 128] = relu(h_fwd||h_bwd)
// blockDim = 128 (4 waves), 64 rows per block, gridDim.y = direction
// ---------------------------------------------------------------------------
__global__ __launch_bounds__(128)
void lstm_kernel(const float* __restrict__ x,
                 const float* __restrict__ WihF, const float* __restrict__ WhhF,
                 const float* __restrict__ WihB, const float* __restrict__ WhhB,
                 float* __restrict__ out, int rows) {
  __shared__ __align__(32) bf16 sWT[256][96];  // W^T: [out col j][k]; k<32 = Wih, k>=32 = Whh
  __shared__ __align__(32) bf16 sX[64][32];
  __shared__ __align__(32) bf16 sH[64][64];

  const int dir = blockIdx.y;
  const float* Wih = dir ? WihB : WihF;
  const float* Whh = dir ? WhhB : WhhF;
  const int row0 = blockIdx.x * 64;
  const int tid = threadIdx.x;

  for (int idx = tid; idx < 256 * 96; idx += 128) {
    int j = idx / 96, k = idx % 96;
    float w = (k < 32) ? Wih[k * 256 + j] : Whh[(k - 32) * 256 + j];
    sWT[j][k] = (bf16)w;
  }
  for (int idx = tid; idx < 64 * 64; idx += 128) (&sH[0][0])[idx] = (bf16)0.0f;
  __syncthreads();

  const int wv = tid >> 5, lane = tid & 31;
  const int col = lane & 15, half = lane >> 4;
  const int mrow = wv * 16;

  float c[4][8], h[4][8];
#pragma unroll
  for (int a = 0; a < 4; ++a)
#pragma unroll
    for (int r = 0; r < 8; ++r) { c[a][r] = 0.0f; h[a][r] = 0.0f; }

  const bf16* WT = &sWT[0][0];

  for (int t = 0; t < TT; ++t) {
    const int tt = dir ? (TT - 1 - t) : t;
    // stage x_t tile
    for (int idx = tid; idx < 64 * 32; idx += 128) {
      int r = idx >> 5, cc = idx & 31;
      int gr = row0 + r;
      float v = (gr < rows) ? x[(size_t)gr * TT * FF + tt * FF + cc] : 0.0f;
      sX[r][cc] = (bf16)v;
    }
    __syncthreads();

    v16bf a0 = afrag(&sX[mrow + col][0], half);          // K 0..31  (x)
    v16bf a1 = afrag(&sH[mrow + col][0], half);          // K 32..63 (h lo)
    v16bf a2 = afrag(&sH[mrow + col][32], half);         // K 64..95 (h hi)

#pragma unroll
    for (int t4 = 0; t4 < 4; ++t4) {
      const int ji = t4 * 16 + col;
      v8f gi = vzero8(), gf = vzero8(), gg = vzero8(), go = vzero8();
      gi = wmma_bf16(a0, bfragW(WT, ji,        0, half, 96), gi);
      gi = wmma_bf16(a1, bfragW(WT, ji,       32, half, 96), gi);
      gi = wmma_bf16(a2, bfragW(WT, ji,       64, half, 96), gi);
      gf = wmma_bf16(a0, bfragW(WT, 64 + ji,   0, half, 96), gf);
      gf = wmma_bf16(a1, bfragW(WT, 64 + ji,  32, half, 96), gf);
      gf = wmma_bf16(a2, bfragW(WT, 64 + ji,  64, half, 96), gf);
      gg = wmma_bf16(a0, bfragW(WT, 128 + ji,  0, half, 96), gg);
      gg = wmma_bf16(a1, bfragW(WT, 128 + ji, 32, half, 96), gg);
      gg = wmma_bf16(a2, bfragW(WT, 128 + ji, 64, half, 96), gg);
      go = wmma_bf16(a0, bfragW(WT, 192 + ji,  0, half, 96), go);
      go = wmma_bf16(a1, bfragW(WT, 192 + ji, 32, half, 96), go);
      go = wmma_bf16(a2, bfragW(WT, 192 + ji, 64, half, 96), go);
#pragma unroll
      for (int r = 0; r < 8; ++r) {
        float cv = sigm(gf[r]) * c[t4][r] + sigm(gi[r]) * fast_tanh(gg[r]);
        c[t4][r] = cv;
        float hv = sigm(go[r]) * fast_tanh(cv);
        h[t4][r] = hv;
        sH[mrow + half * 8 + r][t4 * 16 + col] = (bf16)hv;
      }
    }
    __syncthreads();
  }

#pragma unroll
  for (int t4 = 0; t4 < 4; ++t4)
#pragma unroll
    for (int r = 0; r < 8; ++r) {
      int gr = row0 + mrow + half * 8 + r;
      if (gr < rows)
        out[(size_t)gr * 128 + dir * 64 + t4 * 16 + col] = fmaxf(h[t4][r], 0.0f);
    }
}

// ---------------------------------------------------------------------------
// Generic WMMA GEMM: C[M,128] = act(A[M,K]_bf16 @ W[K,128]_f32 + bias)
// blockDim = 256 (8 waves), 128-row tiles; K multiple of 32
// ---------------------------------------------------------------------------
__global__ __launch_bounds__(256)
void gemm_kernel(const bf16* __restrict__ A, int M, int K,
                 const float* __restrict__ W, const float* __restrict__ bias,
                 float* __restrict__ outF, bf16* __restrict__ outB,
                 int ldc, int do_relu) {
  __shared__ __align__(32) bf16 sA[128][32];
  __shared__ __align__(32) bf16 sWT[128][32];  // W^T chunk: [col j][k 0..31]
  const int row0 = blockIdx.x * 128;
  const int tid = threadIdx.x;
  const int wv = tid >> 5, lane = tid & 31;
  const int col = lane & 15, half = lane >> 4;
  const int mrow = wv * 16;

  v8f acc[8];
#pragma unroll
  for (int n = 0; n < 8; ++n) acc[n] = vzero8();

  for (int kb = 0; kb < K; kb += 32) {
    {  // stage A: 2 threads per row, 16 bf16 each
      int r = tid >> 1, ch = tid & 1;
      int gr = row0 + r;
      bf16* dstp = &sA[r][ch * 16];
      if (gr < M) {
        const v8bf* src = (const v8bf*)(A + (size_t)gr * K + kb + ch * 16);
        *(v8bf*)dstp = src[0];
        *(v8bf*)(dstp + 8) = src[1];
      } else {
#pragma unroll
        for (int i = 0; i < 16; ++i) dstp[i] = (bf16)0.0f;
      }
    }
    {  // stage W^T: 2 threads per output col
      int j = tid >> 1, ch = tid & 1;
#pragma unroll
      for (int i = 0; i < 16; ++i)
        sWT[j][ch * 16 + i] = (bf16)W[(size_t)(kb + ch * 16 + i) * 128 + j];
    }
    __syncthreads();

    v16bf a = afrag(&sA[mrow + col][0], half);
#pragma unroll
    for (int n = 0; n < 8; ++n) {
      v16bf b = bfragW(&sWT[0][0], n * 16 + col, 0, half, 32);
      acc[n] = wmma_bf16(a, b, acc[n]);
    }
    __syncthreads();
  }

#pragma unroll
  for (int n = 0; n < 8; ++n) {
    int cj = n * 16 + col;
    float bv = bias ? bias[cj] : 0.0f;
#pragma unroll
    for (int r = 0; r < 8; ++r) {
      int gr = row0 + mrow + half * 8 + r;
      if (gr < M) {
        float v = acc[n][r] + bv;
        if (do_relu) v = fmaxf(v, 0.0f);
        if (outF) outF[(size_t)gr * ldc + cj] = v;
        if (outB) outB[(size_t)gr * ldc + cj] = (bf16)v;
      }
    }
  }
}

// ---------------------------------------------------------------------------
// Elementwise / graph helpers
// ---------------------------------------------------------------------------
__global__ void fill_kernel(float* __restrict__ p, float v, size_t n) {
  for (size_t i = (size_t)blockIdx.x * blockDim.x + threadIdx.x; i < n;
       i += (size_t)gridDim.x * blockDim.x)
    p[i] = v;
}

__global__ void deg_kernel(const int* __restrict__ dst, const float* __restrict__ w,
                           float* __restrict__ deg, int E) {
  for (int e = blockIdx.x * blockDim.x + threadIdx.x; e < E; e += gridDim.x * blockDim.x)
    __hip_atomic_fetch_add(&deg[dst[e]], w[e], __ATOMIC_RELAXED, __HIP_MEMORY_SCOPE_AGENT);
}

__global__ void rsqrt_kernel(float* __restrict__ d, int n) {
  for (int i = blockIdx.x * blockDim.x + threadIdx.x; i < n; i += gridDim.x * blockDim.x) {
    float v = d[i];
    d[i] = (v > 0.0f) ? rsqrtf(v) : 0.0f;
  }
}

__global__ void concat_kernel(const float* __restrict__ a, const float* __restrict__ b,
                              bf16* __restrict__ o, int rows) {
  size_t n = (size_t)rows * 160;
  for (size_t i = (size_t)blockIdx.x * blockDim.x + threadIdx.x; i < n;
       i += (size_t)gridDim.x * blockDim.x) {
    size_t r = i / 160;
    int j = (int)(i % 160);
    float v = (j < 32) ? a[r * 32 + j] : b[r * 128 + (j - 32)];
    o[i] = (bf16)v;
  }
}

__global__ void gather_kernel(const float* __restrict__ src, const int* __restrict__ idx,
                              float* __restrict__ o) {
  int i = blockIdx.x * blockDim.x + threadIdx.x;
  if (i < BB * 128) {
    int b = i >> 7, j = i & 127;
    o[i] = src[(size_t)idx[b] * 128 + j];
  }
}

// one edge per wave-iteration; lane covers 4 of the 128 feature floats.
// Prefetch the next edge's source row (global_prefetch_b8) — the 52 MB xw
// array is L2-resident (192 MB) but random-access, so hide the gather latency.
__global__ __launch_bounds__(256)
void gcn_agg_kernel(const int* __restrict__ src, const int* __restrict__ dst,
                    const float* __restrict__ w, const float* __restrict__ dinv,
                    const float* __restrict__ xw, float* __restrict__ agg, int E) {
  const int wv = threadIdx.x >> 5, lane = threadIdx.x & 31;
  const int e0 = (blockIdx.x * 8 + wv) * 16;
#pragma unroll 1
  for (int i = 0; i < 16; ++i) {
    int e = e0 + i;
    if (e < E) {
      int s = src[e], d = dst[e];
      int en = e + 1;
      if (en < E)
        __builtin_prefetch(xw + (size_t)src[en] * 128 + lane * 4, 0, 0);
      float coef = dinv[s] * w[e] * dinv[d];
      v4f xs = *(const v4f*)(xw + (size_t)s * 128 + lane * 4);
      float* dp = agg + (size_t)d * 128 + lane * 4;
#pragma unroll
      for (int k = 0; k < 4; ++k)
        __hip_atomic_fetch_add(dp + k, coef * xs[k], __ATOMIC_RELAXED, __HIP_MEMORY_SCOPE_AGENT);
    }
  }
}

// g = relu(agg + dinv^2 * xw_self + bias) -> bf16
__global__ void gcn_epi_kernel(const float* __restrict__ agg, const float* __restrict__ xw,
                               const float* __restrict__ dinv, const float* __restrict__ bias,
                               bf16* __restrict__ o) {
  size_t n = (size_t)NN * 128;
  for (size_t i = (size_t)blockIdx.x * blockDim.x + threadIdx.x; i < n;
       i += (size_t)gridDim.x * blockDim.x) {
    int row = (int)(i >> 7), j = (int)(i & 127);
    float di = dinv[row];
    float v = agg[i] + di * di * xw[i] + bias[j];
    o[i] = (bf16)fmaxf(v, 0.0f);
  }
}

// mean over 25 contiguous nodes per graph
__global__ __launch_bounds__(128)
void pool_kernel(const bf16* __restrict__ g, float* __restrict__ fam, bf16* __restrict__ pc) {
  int b = blockIdx.x, j = threadIdx.x;
  float s = 0.0f;
#pragma unroll 1
  for (int k = 0; k < NN / BB; ++k) s += (float)g[((size_t)b * (NN / BB) + k) * 128 + j];
  s *= (1.0f / (float)(NN / BB));
  fam[(size_t)b * 128 + j] = s;
  pc[(size_t)b * 256 + 128 + j] = (bf16)s;
}

// four sigmoid heads; one wave each
__global__ __launch_bounds__(128)
void heads_kernel(const bf16* __restrict__ combo, const bf16* __restrict__ pc,
                  const float* __restrict__ fam, const float* __restrict__ lgt,
                  const float* Wcom, const float* bcom, const float* Wp, const float* bp,
                  const float* Wfa, const float* bfa, const float* Wls, const float* bls,
                  float* __restrict__ out) {
  int b = blockIdx.x, wv = threadIdx.x >> 5, lane = threadIdx.x & 31;
  float v = 0.0f;
#pragma unroll
  for (int k = 0; k < 4; ++k) {
    int j = lane * 4 + k;
    if (wv == 0)      v += (float)combo[(size_t)b * 128 + j] * Wcom[j];
    else if (wv == 1) v += (float)pc[(size_t)b * 256 + j] * Wp[j];
    else if (wv == 2) v += fam[(size_t)b * 128 + j] * Wfa[j];
    else              v += lgt[(size_t)b * 128 + j] * Wls[j];
  }
#pragma unroll
  for (int off = 16; off > 0; off >>= 1) v += __shfl_down(v, off, 32);
  if (lane == 0) {
    float bias = (wv == 0) ? bcom[0] : (wv == 1) ? bp[0] : (wv == 2) ? bfa[0] : bls[0];
    out[(size_t)wv * BB + b] = sigm(v + bias);
  }
}

// ---------------------------------------------------------------------------
extern "C" void kernel_launch(void* const* d_in, const int* in_sizes, int n_in,
                              void* d_out, int out_size, void* d_ws, size_t ws_size,
                              hipStream_t stream) {
  (void)in_sizes; (void)n_in; (void)out_size; (void)ws_size;
  const float* x_static_node  = (const float*)d_in[0];
  const float* x_static_graph = (const float*)d_in[1];
  const float* x_long_node    = (const float*)d_in[2];
  const float* x_long_graph   = (const float*)d_in[3];
  const int*   edge_index     = (const int*)d_in[4];
  const float* edge_weight    = (const float*)d_in[5];
  const int*   target_index   = (const int*)d_in[7];
  const float* Wih_f = (const float*)d_in[8],  *Whh_f = (const float*)d_in[9];
  const float* Wih_b = (const float*)d_in[10], *Whh_b = (const float*)d_in[11];
  const float* Wl1 = (const float*)d_in[12], *bl1 = (const float*)d_in[13];
  const float* Wl2 = (const float*)d_in[14], *bl2 = (const float*)d_in[15];
  const float* Wc1 = (const float*)d_in[16], *bc1 = (const float*)d_in[17];
  const float* Wc2 = (const float*)d_in[18], *bc2 = (const float*)d_in[19];
  const float* Wpf = (const float*)d_in[20], *bpf = (const float*)d_in[21];
  const float* Wcom = (const float*)d_in[22], *bcom = (const float*)d_in[23];
  const float* Wp  = (const float*)d_in[24], *bp  = (const float*)d_in[25];
  const float* Wfa = (const float*)d_in[26], *bfa = (const float*)d_in[27];
  const float* Wls = (const float*)d_in[28], *bls = (const float*)d_in[29];
  float* out = (float*)d_out;

  char* ws = (char*)d_ws;
  size_t o = 0;
  auto alloc = [&](size_t bytes) { size_t r = o; o += (bytes + 255) & ~(size_t)255; return r; };
  float* longG = (float*)(ws + alloc((size_t)NN * 128 * 4));
  bf16*  xg    = (bf16*)(ws + alloc((size_t)NN * 160 * 2));
  float* xw    = (float*)(ws + alloc((size_t)NN * 128 * 4));
  float* agg   = (float*)(ws + alloc((size_t)NN * 128 * 4));
  bf16*  gbf   = (bf16*)(ws + alloc((size_t)NN * 128 * 2));
  float* deg   = (float*)(ws + alloc((size_t)NN * 4));
  float* lgt   = (float*)(ws + alloc((size_t)BB * 128 * 4));
  float* longN = (float*)(ws + alloc((size_t)BB * 128 * 4));
  bf16*  xp    = (bf16*)(ws + alloc((size_t)BB * 160 * 2));
  bf16*  p1    = (bf16*)(ws + alloc((size_t)BB * 128 * 2));
  bf16*  pc    = (bf16*)(ws + alloc((size_t)BB * 256 * 2));
  bf16*  combo = (bf16*)(ws + alloc((size_t)BB * 128 * 2));
  float* fam   = (float*)(ws + alloc((size_t)BB * 128 * 4));

  auto cdiv = [](size_t a, size_t b) { return (unsigned)((a + b - 1) / b); };

  // 1) bidirectional LSTMs (WMMA-heavy)
  lstm_kernel<<<dim3(NN / 64, 2), 128, 0, stream>>>(x_long_graph, Wih_f, Whh_f, Wih_b, Whh_b, longG, NN);
  lstm_kernel<<<dim3(BB / 64, 2), 128, 0, stream>>>(x_long_node,  Wih_f, Whh_f, Wih_b, Whh_b, longN, BB);

  // 2) concats + target gather
  concat_kernel<<<cdiv((size_t)NN * 160, 256), 256, 0, stream>>>(x_static_graph, longG, xg, NN);
  concat_kernel<<<cdiv((size_t)BB * 160, 256), 256, 0, stream>>>(x_static_node, longN, xp, BB);
  gather_kernel<<<cdiv((size_t)BB * 128, 256), 256, 0, stream>>>(longG, target_index, lgt);

  // 3) symmetric-norm degrees (self loop weight 1)
  fill_kernel<<<cdiv(NN, 256), 256, 0, stream>>>(deg, 1.0f, (size_t)NN);
  deg_kernel<<<cdiv(EE, 256), 256, 0, stream>>>(edge_index + EE, edge_weight, deg, EE);
  rsqrt_kernel<<<cdiv(NN, 256), 256, 0, stream>>>(deg, NN);

  const int* esrc = edge_index;
  const int* edst = edge_index + EE;

  // 4) GCN layer 1
  gemm_kernel<<<NN / 128, 256, 0, stream>>>(xg, NN, 160, Wc1, nullptr, xw, nullptr, 128, 0);
  fill_kernel<<<cdiv((size_t)NN * 128, 256), 256, 0, stream>>>(agg, 0.0f, (size_t)NN * 128);
  gcn_agg_kernel<<<cdiv(EE, 128), 256, 0, stream>>>(esrc, edst, edge_weight, deg, xw, agg, EE);
  gcn_epi_kernel<<<cdiv((size_t)NN * 128, 256), 256, 0, stream>>>(agg, xw, deg, bc1, gbf);

  // 5) GCN layer 2
  gemm_kernel<<<NN / 128, 256, 0, stream>>>(gbf, NN, 128, Wc2, nullptr, xw, nullptr, 128, 0);
  fill_kernel<<<cdiv((size_t)NN * 128, 256), 256, 0, stream>>>(agg, 0.0f, (size_t)NN * 128);
  gcn_agg_kernel<<<cdiv(EE, 128), 256, 0, stream>>>(esrc, edst, edge_weight, deg, xw, agg, EE);
  gcn_epi_kernel<<<cdiv((size_t)NN * 128, 256), 256, 0, stream>>>(agg, xw, deg, bc2, gbf);

  // 6) mean pool -> fam (f32) and pc[:,128:256] (bf16)
  pool_kernel<<<BB, 128, 0, stream>>>(gbf, fam, pc);

  // 7) patient MLP -> pc[:,0:128] (bf16)
  gemm_kernel<<<BB / 128, 256, 0, stream>>>(xp, BB, 160, Wl1, bl1, nullptr, p1, 128, 1);
  gemm_kernel<<<BB / 128, 256, 0, stream>>>(p1, BB, 128, Wl2, bl2, nullptr, pc, 256, 1);

  // 8) combo = relu([patient|fam] @ Wpf + bpf)
  gemm_kernel<<<BB / 128, 256, 0, stream>>>(pc, BB, 256, Wpf, bpf, nullptr, combo, 128, 1);

  // 9) sigmoid heads -> d_out = [out | p_out | f_out | l_out]
  heads_kernel<<<BB, 128, 0, stream>>>(combo, pc, fam, lgt,
                                       Wcom, bcom, Wp, bp, Wfa, bfa, Wls, bls, out);
}
